// MetaEvidentialGSL_31825707664087
// MI455X (gfx1250) — matrix operand order repaired and checked by reading, not compiled, not run on hardware
//
#include <hip/hip_runtime.h>
#include <hip/hip_bf16.h>
#include <math.h>
#include <stdint.h>

// ---------------------------------------------------------------------------
// CDNA5 (gfx1250) implementation: bf16 WMMA GEMMs + fused flash-attention
// with async global->LDS staging (double-buffered, ASYNCcnt-tracked).
// ---------------------------------------------------------------------------

typedef __attribute__((ext_vector_type(16))) __bf16 v16bf;
typedef __attribute__((ext_vector_type(8)))  __bf16 v8bf;
typedef __attribute__((ext_vector_type(8)))  float  v8f;

__device__ __forceinline__ v8f vzero8() {
  v8f z = {0.f,0.f,0.f,0.f,0.f,0.f,0.f,0.f};
  return z;
}

// A-operand (16x32 bf16, MxK) per-lane load, row-major source row pointer.
// Layout: lanes 0-15 (row M=lane) hold K 0..15;
//         lanes 16-31 (row M=lane-16) hold K 16..31 (matches load_b16).
__device__ __forceinline__ v16bf load_a16(const __bf16* rowptr, int half) {
  union { v16bf v; v8bf h[2]; } u;
  u.h[0] = *(const v8bf*)(rowptr + 16 * half);
  u.h[1] = *(const v8bf*)(rowptr + 16 * half + 8);
  return u.v;
}

// B-operand (32x16 bf16, KxN) per-lane load from B^T (row = output column):
// lanes 0-15 hold K 0..15 (contiguous 32B), lanes 16-31 hold K 16..31.
__device__ __forceinline__ v16bf load_b16(const __bf16* p) {
  return *(const v16bf*)p;
}

__device__ __forceinline__ float gelu_exact(float x) {
  return 0.5f * x * (1.0f + erff(x * 0.70710678118654752f));
}
__device__ __forceinline__ float softplus_f(float x) {
  return (x > 20.f) ? x : log1pf(expf(x));
}

// Async global -> LDS copy of one 16B chunk (CDNA5 ASYNCcnt-tracked DMA).
// Inline asm: portable across ROCm-7.2 / amdgpu-toolchain builtin skews.
__device__ __forceinline__ void async_g2l_b128(uint32_t lds_off, const void* g) {
  asm volatile("global_load_async_to_lds_b128 %0, %1, off"
               :: "v"(lds_off), "v"(g) : "memory");
}
__device__ __forceinline__ void wait_asynccnt0() {
  asm volatile("s_wait_asynccnt 0" ::: "memory");
}

// ------------------------------ conversions --------------------------------

__global__ void cvt_bf16_kernel(const float* __restrict__ in,
                                __bf16* __restrict__ out, int n) {
  int i = blockIdx.x * blockDim.x + threadIdx.x;
  if (i < n) out[i] = (__bf16)in[i];
}

// in: R x C (f32, row-major)  ->  out: C x R (bf16, row-major)  == in^T
__global__ void cvt_tr_bf16_kernel(const float* __restrict__ in,
                                   __bf16* __restrict__ out, int R, int C) {
  int i = blockIdx.x * blockDim.x + threadIdx.x;
  if (i < R * C) {
    int r = i / C, c = i - r * C;
    out[(size_t)c * R + r] = (__bf16)in[i];
  }
}

// ------------------------------ WMMA GEMM ----------------------------------
// Out(MxN) = epilogue( A(MxK,bf16) @ B(KxN) + bias ), B given transposed
// (Bt = N x K, row-major bf16). 4 waves / block; each wave owns a 16x64 tile.
// The k-loop is software-pipelined AND unrolled by 2 so the A/B register sets
// ping-pong by role (no v_mov rotation): loads for step i+1 are in flight
// while step i's four WMMAs issue. K must be a multiple of 64 (768/512/256).
__global__ __launch_bounds__(128) void wmma_gemm_kernel(
    const __bf16* __restrict__ A, const __bf16* __restrict__ Bt,
    const float* __restrict__ bias, const float* __restrict__ resid,
    __bf16* __restrict__ Out, int M, int K, int Nout, int act)
{
  const int lane = threadIdx.x & 31;
  const int wave = threadIdx.x >> 5;
  const int r    = lane & 15;
  const int half = lane >> 4;
  const int m0   = blockIdx.x * 64 + wave * 16;
  const int c0   = blockIdx.y * 64;

  v8f acc[4];
  #pragma unroll
  for (int t = 0; t < 4; ++t) acc[t] = vzero8();

  const __bf16* arow = A + (size_t)(m0 + r) * K;
  const __bf16* brow[4];
  #pragma unroll
  for (int t = 0; t < 4; ++t)
    brow[t] = Bt + (size_t)(c0 + 16 * t + r) * K + 16 * half;

  v16bf aA = load_a16(arow, half);
  v16bf bA[4];
  #pragma unroll
  for (int t = 0; t < 4; ++t) bA[t] = load_b16(brow[t]);

  int kk = 0;
  for (; kk + 64 < K; kk += 64) {
    v16bf aB = load_a16(arow + kk + 32, half);
    v16bf bB[4];
    #pragma unroll
    for (int t = 0; t < 4; ++t) bB[t] = load_b16(brow[t] + kk + 32);
    __builtin_prefetch(arow + kk + 96, 0, 1);
    #pragma unroll
    for (int t = 0; t < 4; ++t)
      acc[t] = __builtin_amdgcn_wmma_f32_16x16x32_bf16(
          false, aA, false, bA[t], (short)0, acc[t], false, false);
    aA = load_a16(arow + kk + 64, half);
    #pragma unroll
    for (int t = 0; t < 4; ++t) bA[t] = load_b16(brow[t] + kk + 64);
    #pragma unroll
    for (int t = 0; t < 4; ++t)
      acc[t] = __builtin_amdgcn_wmma_f32_16x16x32_bf16(
          false, aB, false, bB[t], (short)0, acc[t], false, false);
  }
  // Tail: exactly two k-steps remain (kk, kk+32).
  {
    v16bf aB = load_a16(arow + kk + 32, half);
    v16bf bB[4];
    #pragma unroll
    for (int t = 0; t < 4; ++t) bB[t] = load_b16(brow[t] + kk + 32);
    #pragma unroll
    for (int t = 0; t < 4; ++t)
      acc[t] = __builtin_amdgcn_wmma_f32_16x16x32_bf16(
          false, aA, false, bA[t], (short)0, acc[t], false, false);
    #pragma unroll
    for (int t = 0; t < 4; ++t)
      acc[t] = __builtin_amdgcn_wmma_f32_16x16x32_bf16(
          false, aB, false, bB[t], (short)0, acc[t], false, false);
  }

  #pragma unroll
  for (int t = 0; t < 4; ++t) {
    const int col = c0 + 16 * t + r;
    const float bv = bias ? bias[col] : 0.f;
    #pragma unroll
    for (int v = 0; v < 8; ++v) {
      const int row = m0 + v + 8 * half;   // C/D layout: VGPR v <-> row v+8*half
      float x = acc[t][v] + bv;
      if (act) x = gelu_exact(x);
      if (resid) x += resid[(size_t)row * Nout + col];
      Out[(size_t)row * Nout + col] = (__bf16)x;
    }
  }
}

// ------------------------- NIG head (K=256 -> 4) ---------------------------

__global__ void nig_head_kernel(const __bf16* __restrict__ Hf,
                                const float* __restrict__ Wh,
                                const float* __restrict__ bh,
                                const float* __restrict__ gamp,
                                float* __restrict__ outb,
                                float* __restrict__ Gout, int N, int Hdim)
{
  int i = blockIdx.x * blockDim.x + threadIdx.x;
  if (i >= N) return;
  float a0 = bh[0], a1 = bh[1], a2 = bh[2], a3 = bh[3];
  const __bf16* hrow = Hf + (size_t)i * Hdim;
  for (int k = 0; k < Hdim; ++k) {
    float h = (float)hrow[k];
    a0 += h * Wh[k * 4 + 0];
    a1 += h * Wh[k * 4 + 1];
    a2 += h * Wh[k * 4 + 2];
    a3 += h * Wh[k * 4 + 3];
  }
  float mu = a0;
  float vv = softplus_f(a1) + 1e-6f;
  float al = softplus_f(a2) + 1.0f + 1e-6f;
  float be = softplus_f(a3) + 1e-6f;
  outb[i] = mu; outb[N + i] = vv; outb[2 * N + i] = al; outb[3 * N + i] = be;
  if (Gout) {
    float u = be / fmaxf(al - 1.0f, 1e-8f);
    float sg = 1.0f / (1.0f + expf(-u));
    Gout[i] = 1.0f - gamp[0] * sg;
  }
}

// ----------------------------- flash attention -----------------------------
// M_pre[i] = (sum_j e^{s_ij-m_i} G[j] X[j]) / (sum_j e^{s_ij-m_i} G[j])
// (G[i] and the softmax normalizer cancel algebraically -> flash form).
// Block: 128 threads (4 waves) handle 16 query rows. K-blocks (32 x 768 bf16,
// 48 KB) are staged into double-buffered LDS with global_load_async_to_lds
// issued one block ahead; the DMA overlaps S-WMMAs + softmax and is waited
// (ASYNCcnt) right before the barrier that already guards the P' tile.
__global__ __launch_bounds__(128) void flash_attn_kernel(
    const __bf16* __restrict__ Qbf, const __bf16* __restrict__ Kbf,
    const __bf16* __restrict__ Xt,  const float* __restrict__ G,
    __bf16* __restrict__ Mpre, int N, int D, float scale)
{
  __shared__ float  sl[4 * 2 * 8 * 32];              // per-wave S partials (8 KB)
  __shared__ __align__(32) __bf16 pl[16 * 32];       // P' tile for A re-layout
  __shared__ __align__(16) __bf16 kbuf[2][32 * 768]; // K-block double buffer (96 KB)

  const int tid    = threadIdx.x;
  const int lane   = tid & 31;
  const int wave   = tid >> 5;
  const int r      = lane & 15;
  const int half   = lane >> 4;
  const int q0     = blockIdx.x * 16;
  const int kslice = wave * 192;                     // this wave's slice of D

  v8f acc[12];
  #pragma unroll
  for (int i = 0; i < 12; ++i) acc[i] = vzero8();
  float mrow[8], lrow[8];
  #pragma unroll
  for (int v = 0; v < 8; ++v) { mrow[v] = -1e30f; lrow[v] = 0.f; }

  const __bf16* qrow = Qbf + (size_t)(q0 + r) * D;
  const int chunks = 32 * D * 2 / 16;                // 16B chunks per K block

  // Prologue: stage K block 0 into buffer 0.
  {
    uint32_t lbase = (uint32_t)(uintptr_t)&kbuf[0][0];
    const char* gbase = (const char*)Kbf;            // j0 == 0
    for (int c = tid; c < chunks; c += 128)
      async_g2l_b128(lbase + c * 16, gbase + c * 16);
  }
  wait_asynccnt0();
  __syncthreads();

  for (int j0 = 0; j0 < N; j0 += 32) {
    const int buf = (j0 >> 5) & 1;
    // Kick off async staging of the NEXT K block into the other buffer.
    if (j0 + 32 < N) {
      uint32_t lbase = (uint32_t)(uintptr_t)&kbuf[buf ^ 1][0];
      const char* gbase = (const char*)(Kbf + (size_t)(j0 + 32) * D);
      for (int c = tid; c < chunks; c += 128)
        async_g2l_b128(lbase + c * 16, gbase + c * 16);
    }

    // ---- partial S = Q(16 x 192) * K^T over this wave's k-slice (K in LDS) --
    const __bf16* kb = &kbuf[buf][0];
    v8f s0 = vzero8(), s1 = vzero8();
    #pragma unroll
    for (int ks = 0; ks < 6; ++ks) {
      const int kk = kslice + ks * 32;
      v16bf a  = load_a16(qrow + kk, half);
      v16bf b0 = load_b16(kb + (size_t)r * D + kk + 16 * half);
      v16bf b1 = load_b16(kb + (size_t)(16 + r) * D + kk + 16 * half);
      s0 = __builtin_amdgcn_wmma_f32_16x16x32_bf16(
          false, a, false, b0, (short)0, s0, false, false);
      s1 = __builtin_amdgcn_wmma_f32_16x16x32_bf16(
          false, a, false, b1, (short)0, s1, false, false);
    }
    __syncthreads();   // previous iteration's pl readers are done
    #pragma unroll
    for (int v = 0; v < 8; ++v) {
      sl[((wave * 2 + 0) * 8 + v) * 32 + lane] = s0[v];
      sl[((wave * 2 + 1) * 8 + v) * 32 + lane] = s1[v];
    }
    __syncthreads();
    float S0[8], S1[8];
    #pragma unroll
    for (int v = 0; v < 8; ++v) { S0[v] = 0.f; S1[v] = 0.f; }
    #pragma unroll
    for (int w = 0; w < 4; ++w) {
      #pragma unroll
      for (int v = 0; v < 8; ++v) {
        S0[v] += sl[((w * 2 + 0) * 8 + v) * 32 + lane];
        S1[v] += sl[((w * 2 + 1) * 8 + v) * 32 + lane];
      }
    }

    // ---- online softmax (row reductions across 16-lane halves) ----
    const float g0 = G[j0 + r];
    const float g1 = G[j0 + 16 + r];
    float P0[8], P1[8], sc[8];
    #pragma unroll
    for (int v = 0; v < 8; ++v) {
      const float a0 = S0[v] * scale;
      const float a1 = S1[v] * scale;
      float m = fmaxf(a0, a1);
      m = fmaxf(m, __shfl_xor(m, 1));
      m = fmaxf(m, __shfl_xor(m, 2));
      m = fmaxf(m, __shfl_xor(m, 4));
      m = fmaxf(m, __shfl_xor(m, 8));
      const float mnew = fmaxf(mrow[v], m);
      const float s = expf(mrow[v] - mnew);
      sc[v] = s;
      mrow[v] = mnew;
      const float p0 = expf(a0 - mnew) * g0;   // fold G[j] into P'
      const float p1 = expf(a1 - mnew) * g1;
      P0[v] = p0; P1[v] = p1;
      float rs = p0 + p1;
      rs += __shfl_xor(rs, 1);
      rs += __shfl_xor(rs, 2);
      rs += __shfl_xor(rs, 4);
      rs += __shfl_xor(rs, 8);
      lrow[v] = lrow[v] * s + rs;
    }
    #pragma unroll
    for (int i = 0; i < 12; ++i)
      #pragma unroll
      for (int v = 0; v < 8; ++v)
        acc[i][v] *= sc[v];

    // ---- C-layout -> A-layout round trip of P' through LDS ----
    if (wave == 0) {
      #pragma unroll
      for (int v = 0; v < 8; ++v) {
        pl[(v + 8 * half) * 32 + r]      = (__bf16)P0[v];
        pl[(v + 8 * half) * 32 + 16 + r] = (__bf16)P1[v];
      }
    }
    wait_asynccnt0();   // next K block landed (overlapped with work above)
    __syncthreads();    // publishes pl AND kbuf[buf^1] to all waves
    v16bf pa = load_a16(&pl[r * 32], half);

    // ---- O += P'(16x32) @ X(32 x 192slice); B^T rows from Xt (L2-resident),
    //      fully unrolled + pipelined so tile i+1's load overlaps WMMA i ----
    const __bf16* xcol = Xt + (size_t)(kslice + r) * N + j0 + 16 * half;
    v16bf b_cur = load_b16(xcol);
    #pragma unroll
    for (int i = 0; i < 12; ++i) {
      v16bf b_nxt;
      if (i < 11) b_nxt = load_b16(xcol + (size_t)(i + 1) * 16 * N);
      acc[i] = __builtin_amdgcn_wmma_f32_16x16x32_bf16(
          false, pa, false, b_cur, (short)0, acc[i], false, false);
      b_cur = b_nxt;
    }
  }

  // ---- normalize by the gated denominator and store bf16 M_pre ----
  #pragma unroll
  for (int v = 0; v < 8; ++v) {
    const float inv = 1.0f / fmaxf(lrow[v], 1e-20f);
    const int row = q0 + v + 8 * half;
    #pragma unroll
    for (int i = 0; i < 12; ++i) {
      const int col = kslice + i * 16 + r;
      Mpre[(size_t)row * D + col] = (__bf16)(acc[i][v] * inv);
    }
  }
}

// ------------------------------- launcher ----------------------------------

extern "C" void kernel_launch(void* const* d_in, const int* in_sizes, int n_in,
                              void* d_out, int out_size, void* d_ws, size_t ws_size,
                              hipStream_t stream)
{
  const int N = 8192, D = 768, H1 = 512, H2 = 256;

  const float* X    = (const float*)d_in[0];
  const float* Wq   = (const float*)d_in[1];
  const float* Wk   = (const float*)d_in[2];
  const float* Wg   = (const float*)d_in[3];
  const float* bg   = (const float*)d_in[4];
  const float* gam  = (const float*)d_in[5];
  const float* ihW1 = (const float*)d_in[6];
  const float* ihb1 = (const float*)d_in[7];
  const float* ihW2 = (const float*)d_in[8];
  const float* ihb2 = (const float*)d_in[9];
  const float* ihWh = (const float*)d_in[10];
  const float* ihbh = (const float*)d_in[11];
  const float* fhW1 = (const float*)d_in[12];
  const float* fhb1 = (const float*)d_in[13];
  const float* fhW2 = (const float*)d_in[14];
  const float* fhb2 = (const float*)d_in[15];
  const float* fhWh = (const float*)d_in[16];
  const float* fhbh = (const float*)d_in[17];
  float* out = (float*)d_out;

  char* ws = (char*)d_ws;
  size_t off = 0;
  auto take = [&](size_t bytes) -> char* {
    char* p = ws + off;
    off = (off + bytes + 255) & ~(size_t)255;
    return p;
  };
  __bf16* Xbf   = (__bf16*)take((size_t)N * D * 2);
  __bf16* Xt    = (__bf16*)take((size_t)D * N * 2);   // X^T, serves as V^T
  __bf16* Qbf   = (__bf16*)take((size_t)N * D * 2);
  __bf16* Kbf   = (__bf16*)take((size_t)N * D * 2);
  __bf16* Mpre  = (__bf16*)take((size_t)N * D * 2);
  __bf16* Ybf   = (__bf16*)take((size_t)N * D * 2);
  __bf16* H1bf  = (__bf16*)take((size_t)N * H1 * 2);
  __bf16* H2bf  = (__bf16*)take((size_t)N * H2 * 2);
  __bf16* Wqt   = (__bf16*)take((size_t)D * D * 2);
  __bf16* Wkt   = (__bf16*)take((size_t)D * D * 2);
  __bf16* Wgt   = (__bf16*)take((size_t)D * D * 2);
  __bf16* iW1t  = (__bf16*)take((size_t)D * H1 * 2);
  __bf16* iW2t  = (__bf16*)take((size_t)H1 * H2 * 2);
  __bf16* fW1t  = (__bf16*)take((size_t)D * H1 * 2);
  __bf16* fW2t  = (__bf16*)take((size_t)H1 * H2 * 2);
  float*  Gbuf  = (float*)take((size_t)N * 4);

  const int nXD = N * D;
  const int TB = 256;
  auto cdiv = [](int a, int b) { return (a + b - 1) / b; };

  // --- format conversion / weight transposes (bf16) ---
  cvt_bf16_kernel<<<cdiv(nXD, TB), TB, 0, stream>>>(X, Xbf, nXD);
  cvt_tr_bf16_kernel<<<cdiv(nXD, TB), TB, 0, stream>>>(X, Xt, N, D);
  cvt_tr_bf16_kernel<<<cdiv(D * D, TB), TB, 0, stream>>>(Wq, Wqt, D, D);
  cvt_tr_bf16_kernel<<<cdiv(D * D, TB), TB, 0, stream>>>(Wk, Wkt, D, D);
  cvt_tr_bf16_kernel<<<cdiv(D * D, TB), TB, 0, stream>>>(Wg, Wgt, D, D);
  cvt_tr_bf16_kernel<<<cdiv(D * H1, TB), TB, 0, stream>>>(ihW1, iW1t, D, H1);
  cvt_tr_bf16_kernel<<<cdiv(H1 * H2, TB), TB, 0, stream>>>(ihW2, iW2t, H1, H2);
  cvt_tr_bf16_kernel<<<cdiv(D * H1, TB), TB, 0, stream>>>(fhW1, fW1t, D, H1);
  cvt_tr_bf16_kernel<<<cdiv(H1 * H2, TB), TB, 0, stream>>>(fhW2, fW2t, H1, H2);

  // --- head 1: H1 = gelu(X W1 + b1); H2 = gelu(H1 W2 + b2); head -> m0..b0, G
  wmma_gemm_kernel<<<dim3(N / 64, H1 / 64), 128, 0, stream>>>(
      Xbf, iW1t, ihb1, nullptr, H1bf, N, D, H1, 1);
  wmma_gemm_kernel<<<dim3(N / 64, H2 / 64), 128, 0, stream>>>(
      H1bf, iW2t, ihb2, nullptr, H2bf, N, H1, H2, 1);
  nig_head_kernel<<<cdiv(N, 128), 128, 0, stream>>>(
      H2bf, ihWh, ihbh, gam, out, Gbuf, N, H2);

  // --- Q, K projections ---
  wmma_gemm_kernel<<<dim3(N / 64, D / 64), 128, 0, stream>>>(
      Xbf, Wqt, nullptr, nullptr, Qbf, N, D, D, 0);
  wmma_gemm_kernel<<<dim3(N / 64, D / 64), 128, 0, stream>>>(
      Xbf, Wkt, nullptr, nullptr, Kbf, N, D, D, 0);

  // --- fused gated attention -> M_pre (the G/softmax normalizers cancel) ---
  const float scale = 1.0f / sqrtf((float)D);
  flash_attn_kernel<<<N / 16, 128, 0, stream>>>(
      Qbf, Kbf, Xt, Gbuf, Mpre, N, D, scale);

  // --- Y = X + gelu(M_pre Wg + bg) ---
  wmma_gemm_kernel<<<dim3(N / 64, D / 64), 128, 0, stream>>>(
      Mpre, Wgt, bg, X, Ybf, N, D, D, 1);

  // --- head 2 on Y -> mu, v, al, be ---
  wmma_gemm_kernel<<<dim3(N / 64, H1 / 64), 128, 0, stream>>>(
      Ybf, fW1t, fhb1, nullptr, H1bf, N, D, H1, 1);
  wmma_gemm_kernel<<<dim3(N / 64, H2 / 64), 128, 0, stream>>>(
      H1bf, fW2t, fhb2, nullptr, H2bf, N, H1, H2, 1);
  nig_head_kernel<<<cdiv(N, 128), 128, 0, stream>>>(
      H2bf, fhWh, fhbh, gam, out + 4 * N, nullptr, N, H2);
}
